// Qwen3OmniMoeTransformerDecoderLayer_26663156973646
// MI455X (gfx1250) — matrix operand order repaired and checked by reading, not compiled, not run on hardware
//
#include <hip/hip_runtime.h>

// ---------------------------------------------------------------------------
// Qwen3-Omni MoE decoder MLP for MI455X (gfx1250, wave32, WMMA)
//   T=4096 tokens, H=2048, I=768, E=16, top-2.
// Exact fp32 router + top-2 grouping; sparse expert GEMMs on
// v_wmma_f32_16x16x32_bf16 with 32-token M tiles (2x B-fragment reuse).
// Weights/activations pre-converted to bf16 once per launch when ws allows
// (streaming pass; bf16 weights fit in the 192MB L2).
// ---------------------------------------------------------------------------

#define T_TOK  4096
#define HDIM   2048
#define EDIM   16
#define IDIM   768
#define MTILE  32
#define IPAD   776    // 768 + 8 bf16 pad
#define XPAD   2056   // 2048 + 8 bf16 pad

typedef __attribute__((ext_vector_type(16))) __bf16 bf16x16;
typedef __attribute__((ext_vector_type(8)))  __bf16 bf16x8;
typedef __attribute__((ext_vector_type(8)))  float  f32x8;

__device__ __forceinline__ f32x8 wmma_bf16(bf16x16 a, bf16x16 b, f32x8 c) {
  return __builtin_amdgcn_wmma_f32_16x16x32_bf16(
      false, a, false, b, (short)0, c, false, false);
}

// ---- B fragment: K = off+0..7 and K = off+16..23 per lane -------------------
__device__ __forceinline__ bf16x16 loadB_frag(const float* __restrict__ p) {
  const float4* q = (const float4*)p;
  float4 f0 = q[0], f1 = q[1], f2 = q[4], f3 = q[5];
  bf16x16 r;
  r[0]=(__bf16)f0.x; r[1]=(__bf16)f0.y; r[2]=(__bf16)f0.z; r[3]=(__bf16)f0.w;
  r[4]=(__bf16)f1.x; r[5]=(__bf16)f1.y; r[6]=(__bf16)f1.z; r[7]=(__bf16)f1.w;
  r[8]=(__bf16)f2.x; r[9]=(__bf16)f2.y; r[10]=(__bf16)f2.z; r[11]=(__bf16)f2.w;
  r[12]=(__bf16)f3.x; r[13]=(__bf16)f3.y; r[14]=(__bf16)f3.z; r[15]=(__bf16)f3.w;
  return r;
}
__device__ __forceinline__ bf16x16 loadB_frag(const __bf16* __restrict__ p) {
  bf16x8 lo = *(const bf16x8*)p;
  bf16x8 hi = *(const bf16x8*)(p + 16);
  bf16x16 r;
  #pragma unroll
  for (int j = 0; j < 8; ++j) { r[j] = lo[j]; r[8 + j] = hi[j]; }
  return r;
}

// ---- A fragment from an LDS row: K = off+0..7 and K = off+16..23 ------------
__device__ __forceinline__ bf16x16 ldsA_frag(const __bf16* row, int off) {
  bf16x8 lo = *(const bf16x8*)(row + off);
  bf16x8 hi = *(const bf16x8*)(row + off + 16);
  bf16x16 r;
  #pragma unroll
  for (int j = 0; j < 8; ++j) { r[j] = lo[j]; r[8 + j] = hi[j]; }
  return r;
}

// ---- stage 128 contiguous elements into LDS as bf16 -------------------------
__device__ __forceinline__ void stage128(const float* __restrict__ s, __bf16* d) {
  #pragma unroll
  for (int i = 0; i < 16; ++i) {
    const float4* q = (const float4*)(s + i * 8);
    float4 a = q[0], b = q[1];
    bf16x8 r;
    r[0]=(__bf16)a.x; r[1]=(__bf16)a.y; r[2]=(__bf16)a.z; r[3]=(__bf16)a.w;
    r[4]=(__bf16)b.x; r[5]=(__bf16)b.y; r[6]=(__bf16)b.z; r[7]=(__bf16)b.w;
    *(bf16x8*)(d + i * 8) = r;
  }
}
__device__ __forceinline__ void stage128(const __bf16* __restrict__ s, __bf16* d) {
  #pragma unroll
  for (int i = 0; i < 16; ++i)
    *(bf16x8*)(d + i * 8) = *(const bf16x8*)(s + i * 8);
}

// ---------------------------------------------------------------------------
// fp32 -> bf16 streaming conversion (8 elems / thread)
// ---------------------------------------------------------------------------
__global__ __launch_bounds__(256) void cvt_bf16_kernel(
    const float* __restrict__ src, __bf16* __restrict__ dst, int n8) {
  int i = blockIdx.x * 256 + threadIdx.x;
  if (i >= n8) return;
  const float4* q = (const float4*)src + (size_t)i * 2;
  float4 a = q[0], b = q[1];
  bf16x8 r;
  r[0]=(__bf16)a.x; r[1]=(__bf16)a.y; r[2]=(__bf16)a.z; r[3]=(__bf16)a.w;
  r[4]=(__bf16)b.x; r[5]=(__bf16)b.y; r[6]=(__bf16)b.z; r[7]=(__bf16)b.w;
  ((bf16x8*)dst)[i] = r;
}

// ---------------------------------------------------------------------------
// Router: exact fp32 logits, softmax, top-2, scatter token->expert lists.
// ---------------------------------------------------------------------------
__global__ __launch_bounds__(256) void moe_router_kernel(
    const float* __restrict__ x, const float* __restrict__ gw,
    float* __restrict__ logitsOut, float* __restrict__ probs,
    float* __restrict__ combine, int* __restrict__ cnt,
    int* __restrict__ lists) {
  const int tid = threadIdx.x;
  const int tl  = tid >> 4;
  const int e   = tid & 15;
  const int t   = blockIdx.x * 16 + tl;

  const float* xr = x + (size_t)t * HDIM;
  float acc = 0.f;
  #pragma unroll 4
  for (int h = 0; h < HDIM; h += 4) {
    acc += xr[h]     * gw[(h)     * EDIM + e];
    acc += xr[h + 1] * gw[(h + 1) * EDIM + e];
    acc += xr[h + 2] * gw[(h + 2) * EDIM + e];
    acc += xr[h + 3] * gw[(h + 3) * EDIM + e];
  }

  __shared__ float lg[16][16];
  lg[tl][e] = acc;
  __syncthreads();

  if (e == 0) {
    float row[EDIM], p[EDIM];
    float mx = -3.4e38f;
    #pragma unroll
    for (int i = 0; i < EDIM; ++i) { row[i] = lg[tl][i]; mx = fmaxf(mx, row[i]); }
    float sum = 0.f;
    #pragma unroll
    for (int i = 0; i < EDIM; ++i) { p[i] = __expf(row[i] - mx); sum += p[i]; }
    const float inv = 1.f / sum;
    #pragma unroll
    for (int i = 0; i < EDIM; ++i) p[i] *= inv;

    int i0 = 0;
    #pragma unroll
    for (int i = 1; i < EDIM; ++i) if (p[i] > p[i0]) i0 = i;
    int i1 = (i0 == 0) ? 1 : 0;
    #pragma unroll
    for (int i = 0; i < EDIM; ++i) if (i != i1 && i != i0 && p[i] > p[i1]) i1 = i;
    float w0 = p[i0], w1 = p[i1];
    const float winv = 1.f / (w0 + w1);
    w0 *= winv; w1 *= winv;

    #pragma unroll
    for (int i = 0; i < EDIM; ++i) {
      logitsOut[(size_t)t * EDIM + i] = row[i];
      probs[(size_t)t * EDIM + i]     = p[i];
      combine[(size_t)t * EDIM + i]   = (i == i0) ? w0 : ((i == i1) ? w1 : 0.f);
    }
    int p0 = atomicAdd(&cnt[i0], 1); lists[i0 * T_TOK + p0] = t;
    int p1 = atomicAdd(&cnt[i1], 1); lists[i1 * T_TOK + p1] = t;
  }
}

// ---------------------------------------------------------------------------
// Aux loss: deterministic fixed-order reduction.
// ---------------------------------------------------------------------------
__global__ void moe_aux_kernel(const float* __restrict__ probs,
                               const float* __restrict__ combine,
                               float* __restrict__ auxOut) {
  __shared__ float ps[EDIM], fs[EDIM];
  const int e = threadIdx.x;
  if (e < EDIM) {
    float sp = 0.f, sf = 0.f;
    for (int t = 0; t < T_TOK; ++t) {
      sp += probs[(size_t)t * EDIM + e];
      sf += (combine[(size_t)t * EDIM + e] != 0.f) ? 1.f : 0.f;
    }
    ps[e] = sp; fs[e] = sf;
  }
  __syncthreads();
  if (e == 0) {
    float a = 0.f;
    #pragma unroll
    for (int i = 0; i < EDIM; ++i)
      a += (ps[i] / (float)T_TOK) * (fs[i] / (float)T_TOK);
    *auxOut = a * (float)EDIM * 0.001f;
  }
}

// ---------------------------------------------------------------------------
// Fused expert kernel, 32-token M tile. WT = __bf16 (fast) or float (fallback).
// ---------------------------------------------------------------------------
template <typename WT>
__global__ __launch_bounds__(256) void moe_expert_kernel(
    const WT* __restrict__ xw, const WT* __restrict__ gup,
    const WT* __restrict__ down, const float* __restrict__ combine,
    const int* __restrict__ cnt, const int* __restrict__ lists,
    float* __restrict__ out) {
  const int e    = blockIdx.y;
  const int tile = blockIdx.x;
  const int nTok = cnt[e];
  const int t0   = tile * MTILE;
  if (t0 >= nTok) return;

  __shared__ int    tokLds[MTILE];
  __shared__ float  wLds[MTILE];
  __shared__ __bf16 xTile[MTILE][XPAD];
  __shared__ __bf16 interLds[MTILE][IPAD];

  const int tid = threadIdx.x;
  if (tid < MTILE) {
    const int idx = t0 + tid;
    const int tok = (idx < nTok) ? lists[e * T_TOK + idx] : 0;
    tokLds[tid] = tok;
    wLds[tid]   = (idx < nTok) ? combine[(size_t)tok * EDIM + e] : 0.f;
  }
  __syncthreads();

  // ---- stage 32-token x tile into LDS (bf16), 256 elems / thread ----
  {
    const int srow   = tid >> 3;            // 0..31
    const int schunk = (tid & 7) * 256;     // 0..1792
    const WT* src = xw + (size_t)tokLds[srow] * HDIM + schunk;
    stage128(src,       &xTile[srow][schunk]);
    stage128(src + 128, &xTile[srow][schunk + 128]);
  }
  __syncthreads();

  const int wave   = tid >> 5;
  const int lane   = tid & 31;
  const int rowA   = lane & 15;
  const int kbaseA = (lane >> 4) * 8;
  const int colB   = lane & 15;
  const int kbaseB = (lane >> 4) * 8;
  const int rowC   = (lane >> 4) * 8;

  const WT* __restrict__ W1 = gup  + (size_t)e * (2 * IDIM) * HDIM;
  const WT* __restrict__ W2 = down + (size_t)e * HDIM * IDIM;

  // ---- GEMM1: 48 gate/up tile pairs (6 / wave), 2x2 register blocking ----
  for (int pp = 0; pp < 6; ++pp) {
    const int nt = wave * 6 + pp;
    const size_t mg = (size_t)(nt * 16 + colB) * HDIM;
    const size_t mu = (size_t)(nt * 16 + colB + IDIM) * HDIM;
    f32x8 cg0 = {}, cg1 = {}, cu0 = {}, cu1 = {};
    for (int h0 = 0; h0 < HDIM; h0 += 32) {
      bf16x16 a0 = ldsA_frag(&xTile[rowA][0],      h0 + kbaseA);
      bf16x16 a1 = ldsA_frag(&xTile[rowA + 16][0], h0 + kbaseA);
      bf16x16 bg = loadB_frag(W1 + mg + h0 + kbaseB);
      bf16x16 bu = loadB_frag(W1 + mu + h0 + kbaseB);
      cg0 = wmma_bf16(a0, bg, cg0);
      cg1 = wmma_bf16(a1, bg, cg1);
      cu0 = wmma_bf16(a0, bu, cu0);
      cu1 = wmma_bf16(a1, bu, cu1);
    }
    #pragma unroll
    for (int r = 0; r < 8; ++r) {
      const float g0 = cg0[r], u0 = cu0[r];
      const float g1 = cg1[r], u1 = cu1[r];
      const float s0 = g0 * __builtin_amdgcn_rcpf(1.f + __expf(-g0));
      const float s1 = g1 * __builtin_amdgcn_rcpf(1.f + __expf(-g1));
      interLds[rowC + r][nt * 16 + colB]      = (__bf16)(s0 * u0);
      interLds[rowC + 16 + r][nt * 16 + colB] = (__bf16)(s1 * u1);
    }
  }
  __syncthreads();

  // ---- GEMM2: 128 H tiles (16 / wave), 2x2 register blocking, A from LDS ----
  for (int pp = 0; pp < 8; ++pp) {
    const int nt0 = wave * 16 + pp * 2;
    const int h0c = nt0 * 16 + colB;
    const int h1c = h0c + 16;
    f32x8 c00 = {}, c01 = {}, c10 = {}, c11 = {};
    for (int m0 = 0; m0 < IDIM; m0 += 32) {
      bf16x16 a0 = ldsA_frag(&interLds[rowA][0],      m0 + kbaseA);
      bf16x16 a1 = ldsA_frag(&interLds[rowA + 16][0], m0 + kbaseA);
      bf16x16 b0 = loadB_frag(W2 + (size_t)h0c * IDIM + m0 + kbaseB);
      bf16x16 b1 = loadB_frag(W2 + (size_t)h1c * IDIM + m0 + kbaseB);
      c00 = wmma_bf16(a0, b0, c00);
      c10 = wmma_bf16(a1, b0, c10);
      c01 = wmma_bf16(a0, b1, c01);
      c11 = wmma_bf16(a1, b1, c11);
    }
    #pragma unroll
    for (int r = 0; r < 8; ++r) {
      const int tr0 = rowC + r;
      const int tr1 = rowC + 16 + r;
      if (t0 + tr0 < nTok) {
        float* dst = &out[(size_t)tokLds[tr0] * HDIM];
        atomicAdd(dst + h0c, wLds[tr0] * c00[r]);
        atomicAdd(dst + h1c, wLds[tr0] * c01[r]);
      }
      if (t0 + tr1 < nTok) {
        float* dst = &out[(size_t)tokLds[tr1] * HDIM];
        atomicAdd(dst + h0c, wLds[tr1] * c10[r]);
        atomicAdd(dst + h1c, wLds[tr1] * c11[r]);
      }
    }
  }
}

// ---------------------------------------------------------------------------
extern "C" void kernel_launch(void* const* d_in, const int* in_sizes, int n_in,
                              void* d_out, int out_size, void* d_ws, size_t ws_size,
                              hipStream_t stream) {
  const float* x   = (const float*)d_in[0];   // (4,1024,2048)
  const float* gw  = (const float*)d_in[1];   // (2048,16)
  const float* gup = (const float*)d_in[2];   // (16,1536,2048)
  const float* dwn = (const float*)d_in[3];   // (16,2048,768)

  float* finalOut  = (float*)d_out;
  float* logitsOut = finalOut + (size_t)T_TOK * HDIM;
  float* auxOut    = logitsOut + (size_t)T_TOK * EDIM;

  char*  ws      = (char*)d_ws;
  float* probs   = (float*)(ws);                 // 256 KB
  float* combine = (float*)(ws + 262144);        // 256 KB
  int*   cnt     = (int*)  (ws + 524288);        // 64 B
  int*   lists   = (int*)  (ws + 524352);        // 256 KB

  const size_t N_X  = (size_t)T_TOK * HDIM;            // 8.4M
  const size_t N_W1 = (size_t)EDIM * 2 * IDIM * HDIM;  // 50.3M
  const size_t N_W2 = (size_t)EDIM * HDIM * IDIM;      // 25.2M
  const size_t OFF_XB  = 1u << 20;
  const size_t OFF_W1B = OFF_XB  + N_X  * 2;
  const size_t OFF_W2B = OFF_W1B + N_W1 * 2;
  const size_t NEEDED  = OFF_W2B + N_W2 * 2;

  hipMemsetAsync(finalOut, 0, (size_t)T_TOK * HDIM * sizeof(float), stream);
  hipMemsetAsync(cnt, 0, EDIM * sizeof(int), stream);

  moe_router_kernel<<<T_TOK / 16, 256, 0, stream>>>(x, gw, logitsOut, probs,
                                                    combine, cnt, lists);
  moe_aux_kernel<<<1, 64, 0, stream>>>(probs, combine, auxOut);

  dim3 grid(T_TOK / MTILE, EDIM);  // tiles fastest -> same-expert blocks adjacent

  if (ws_size >= NEEDED) {
    __bf16* xb  = (__bf16*)(ws + OFF_XB);
    __bf16* w1b = (__bf16*)(ws + OFF_W1B);
    __bf16* w2b = (__bf16*)(ws + OFF_W2B);
    cvt_bf16_kernel<<<(int)(N_X  / 8 / 256), 256, 0, stream>>>(x,   xb,  (int)(N_X  / 8));
    cvt_bf16_kernel<<<(int)(N_W1 / 8 / 256), 256, 0, stream>>>(gup, w1b, (int)(N_W1 / 8));
    cvt_bf16_kernel<<<(int)(N_W2 / 8 / 256), 256, 0, stream>>>(dwn, w2b, (int)(N_W2 / 8));
    moe_expert_kernel<__bf16><<<grid, 256, 0, stream>>>(
        xb, w1b, w2b, combine, cnt, lists, finalOut);
  } else {
    moe_expert_kernel<float><<<grid, 256, 0, stream>>>(
        x, gup, dwn, combine, cnt, lists, finalOut);
  }
}
